// SemGCN_59356448030851
// MI455X (gfx1250) — matrix operand order repaired and checked by reading, not compiled, not run on hardware
//
#include <hip/hip_runtime.h>
#include <stdint.h>
#include <stddef.h>

// ---------------------------------------------------------------------------
// SemGCN forward for MI455X (gfx1250, wave32, WMMA).
// bf16 activations, f32 WMMA accumulation.
// GEMM: per-K-step preload of 8 B fragments -> bursts of 8 back-to-back WMMAs
// with a single DS wait; weights staged in LDS via global_load_async_to_lds
// (ASYNCcnt-tracked async copy).  BN stats are a separate bandwidth pass.
// Gconv biases are dropped: they cancel exactly under training-mode BatchNorm.
// Workspace use: ~417 MB (3 bf16 activation buffers + padded input + packed W).
// ---------------------------------------------------------------------------

#define JNT   16
#define HIDC  128
#define CINC  5
#define COUTC 3
#define NEGV  (-9.0e15f)
#define EPSV  (1e-5f)

typedef __attribute__((ext_vector_type(16))) __bf16 v16bf;
typedef __attribute__((ext_vector_type(2)))  __bf16 v2bf;
typedef __attribute__((ext_vector_type(8)))  float  v8f;

union BF16x16 { uint4 u[2]; v16bf v; };

__device__ __forceinline__ uint32_t pk2(float a, float b) {   // 2xf32 -> packed bf16
  v2bf p; p[0] = (__bf16)a; p[1] = (__bf16)b;
  uint32_t r; __builtin_memcpy(&r, &p, 4); return r;
}
__device__ __forceinline__ float bf2f(uint32_t b) {           // bf16 -> f32
  return __uint_as_float((b & 0xFFFFu) << 16);
}

// ---------------------------------------------------------------------------
// Prep: row-softmax of masked logits for all 10 layers -> Mcat[16,32]
// (cols 0..15 = diag branch for W0, cols 16..31 = off-diag for W1).
// Stored raw (f32) and packed TRANSPOSED as WMMA B fragments in this code's
// contiguous-k order (B = Mcat^T [32,16]: lane col i = l%16, element e -> K row h*16+e).
// Also zeroes the BN stats accumulator.
// ---------------------------------------------------------------------------
__global__ void prep_kernel(const float* __restrict__ adj,
                            const float* __restrict__ e_in,
                            const float* __restrict__ e_res,
                            const float* __restrict__ e_out,
                            float* __restrict__ mcatRaw,      // [10][16][32]
                            uint32_t* __restrict__ mcatT,     // [10][32 lanes][8 dw]
                            float* __restrict__ stats) {      // [256]
  int t = threadIdx.x;                                        // block 512
  if (t < 256) stats[t] = 0.0f;
  if (t < 10 * JNT) {
    int L = t / JNT, i = t % JNT;
    const float* e = (L == 0) ? e_in : ((L == 9) ? e_out : (e_res + (L - 1) * JNT * JNT));
    float row[JNT];
    float mx = -3.0e38f;
    for (int j = 0; j < JNT; ++j) {
      float a = adj[i * JNT + j];
      float lg = a * e[i * JNT + j] + (1.0f - a) * NEGV;
      row[j] = lg; mx = fmaxf(mx, lg);
    }
    float s = 0.0f;
    for (int j = 0; j < JNT; ++j) { row[j] = __expf(row[j] - mx); s += row[j]; }
    float inv = 1.0f / s;
    float diag = row[i] * inv;
    float* o = mcatRaw + (L * JNT + i) * 32;
    for (int j = 0; j < JNT; ++j) o[j]      = (j == i) ? diag : 0.0f;
    for (int j = 0; j < JNT; ++j) o[16 + j] = (j == i) ? 0.0f : row[j] * inv;
  }
  __syncthreads();
  if (t < 10 * 32) {                          // pack Mcat^T as B fragment
    int L = t / 32, lane = t % 32, i = lane & 15, h = lane >> 4;
    const float* src = mcatRaw + (L * JNT + i) * 32;   // Mcat row i (j' = K dim)
    uint32_t* dst = mcatT + (size_t)t * 8;
    for (int p = 0; p < 8; ++p)
      dst[p] = pk2(src[h * 16 + 2 * p], src[h * 16 + 2 * p + 1]);
  }
}

// ---------------------------------------------------------------------------
// Pack weights into bf16 WMMA B fragments.
//  Hidden: W_res[8][2][128][128] -> pkRes[8][kk=4][nt=16][lane=32][8 dw]
//  Input : W_in [2][5][128], K padded 5->32 -> pkIn[nt=16][lane=32][8 dw]
//  B frag (32x16): lane (n = l%16, h = l/16), element e -> K = h*16 + e.
// ---------------------------------------------------------------------------
__global__ void pack_kernel(const float* __restrict__ W_in,
                            const float* __restrict__ W_res,
                            uint32_t* __restrict__ pkIn,
                            uint32_t* __restrict__ pkRes) {
  int t = blockIdx.x * blockDim.x + threadIdx.x;
  if (t < 8 * 4 * 16 * 32) {
    int lane = t & 31, nt = (t >> 5) & 15, kk = (t >> 9) & 3, L = t >> 11;
    int n = lane & 15, h = lane >> 4;
    int col = nt * 16 + n, br = col >> 7, c = col & 127;
    uint32_t* dst = pkRes + (size_t)t * 8;
    for (int p = 0; p < 8; ++p) {
      int k0 = kk * 32 + h * 16 + 2 * p;
      float v0 = W_res[((size_t)(L * 2 + br) * 128 + k0) * 128 + c];
      float v1 = W_res[((size_t)(L * 2 + br) * 128 + k0 + 1) * 128 + c];
      dst[p] = pk2(v0, v1);
    }
  } else {
    int t2 = t - 8 * 4 * 16 * 32;
    if (t2 < 16 * 32) {
      int lane = t2 & 31, nt = t2 >> 5;
      int n = lane & 15, h = lane >> 4;
      int col = nt * 16 + n, br = col >> 7, c = col & 127;
      uint32_t* dst = pkIn + (size_t)t2 * 8;
      for (int p = 0; p < 8; ++p) {
        int k0 = h * 16 + 2 * p, k1 = k0 + 1;
        float v0 = (k0 < CINC) ? W_in[((size_t)br * CINC + k0) * 128 + c] : 0.0f;
        float v1 = (k1 < CINC) ? W_in[((size_t)br * CINC + k1) * 128 + c] : 0.0f;
        dst[p] = pk2(v0, v1);
      }
    }
  }
}

// ---------------------------------------------------------------------------
// Pad input x[B,16,5] f32 -> Xp[B,16,32] bf16 (zeros beyond CIN).
// ---------------------------------------------------------------------------
__global__ void pad_kernel(const float* __restrict__ x,
                           uint32_t* __restrict__ Xp, int nwords) {
  int w = blockIdx.x * blockDim.x + threadIdx.x;
  if (w >= nwords) return;
  int r = w >> 4, p = w & 15;
  int k0 = 2 * p, k1 = k0 + 1;
  float v0 = (k0 < CINC) ? x[(size_t)r * CINC + k0] : 0.0f;
  float v1 = (k1 < CINC) ? x[(size_t)r * CINC + k1] : 0.0f;
  Xp[w] = pk2(v0, v1);
}

// ---------------------------------------------------------------------------
// Core WMMA gconv kernel.  One wave handles one batch element (16-row tile):
//   Z[16,256]   = X[16,K] @ [W0|W1]                (16*KSTEPS bf16 WMMAs)
//   Zs^T[128,32]= branch-stacked Z, transposed in per-wave LDS scratch
//   out^T       = Zs^T-tile[16,32] @ Mcat^T[32,16] (8 bf16 WMMAs)
// Weights staged in LDS via async global->LDS copy (ASYNCcnt).  Per K-step,
// all 8 B fragments are preloaded into distinct registers so the 8 WMMAs of
// a batch issue back-to-back behind a single DS wait.  out^T lane layout
// gives 8 consecutive features per lane -> coalesced b128 Y stores.
// ---------------------------------------------------------------------------
template <int KSTEPS>
__global__ __launch_bounds__(128) void gemm_kernel(
    const uint4* __restrict__ Xu,        // bf16 act, row-major [B*16, KSTEPS*32]
    const uint4* __restrict__ Wp,        // packed B fragments for this layer
    const uint4* __restrict__ mcatT,     // packed Mcat^T B fragment (32x2 uint4)
    uint16_t* __restrict__ Y,            // bf16 out [B*16, 128]
    int nTiles) {
  extern __shared__ char smem[];
  uint4* wl = (uint4*)smem;                                    // KSTEPS*1024 uint4
  uint16_t* zsAll = (uint16_t*)(smem + (size_t)KSTEPS * 16384);
  const int WL = KSTEPS * 16 * 32 * 2;
  // Async DMA of packed weights into LDS (no VGPR round-trip, ASYNCcnt).
  for (int i = threadIdx.x; i < WL; i += blockDim.x) {
    uint32_t lds = (uint32_t)(size_t)(wl + i);     // low 32 bits = LDS offset
    unsigned long long ga = (unsigned long long)(size_t)(Wp + i);
    asm volatile("global_load_async_to_lds_b128 %0, %1, off"
                 :: "v"(lds), "v"(ga) : "memory");
  }
  asm volatile("s_wait_asynccnt 0x0" ::: "memory");
  __syncthreads();

  const int lane = threadIdx.x & 31, wave = threadIdx.x >> 5;
  const int m = lane & 15, h = lane >> 4;
  uint16_t* zs = zsAll + (size_t)wave * (128 * 32);            // [d=128][j'=32] bf16

  BF16x16 mct;
  mct.u[0] = mcatT[lane * 2];
  mct.u[1] = mcatT[lane * 2 + 1];

  const int K8 = KSTEPS * 4;                                   // uint4 per row
  const int waveGlobal = blockIdx.x * (blockDim.x >> 5) + wave;
  const int nWaves = gridDim.x * (blockDim.x >> 5);

  for (int b = waveGlobal; b < nTiles; b += nWaves) {
    // software prefetch of the next tile's activation rows
    int bnext = b + nWaves;
    if (bnext < nTiles) {
      const uint4* xn = Xu + (size_t)(bnext * 16 + m) * K8;
#pragma unroll
      for (int q = 0; q < K8; q += 8) __builtin_prefetch(xn + q, 0, 0);
    }
    // --- A fragments: lane m holds row m; K = h*16 .. h*16+15
    BF16x16 afr[KSTEPS];
    const uint4* xrow = Xu + (size_t)(b * 16 + m) * K8;
#pragma unroll
    for (int kk = 0; kk < KSTEPS; ++kk) {
      afr[kk].u[0] = xrow[kk * 4 + 2 * h];
      afr[kk].u[1] = xrow[kk * 4 + 2 * h + 1];
    }
    // --- main GEMM: two batches of 8 N-tiles, 8 independent accumulators
#pragma unroll
    for (int half = 0; half < 2; ++half) {
      v8f acc[8];
#pragma unroll
      for (int n8i = 0; n8i < 8; ++n8i)
        acc[n8i] = (v8f){0.f, 0.f, 0.f, 0.f, 0.f, 0.f, 0.f, 0.f};
#pragma unroll
      for (int kk = 0; kk < KSTEPS; ++kk) {
        BF16x16 bfr[8];                             // distinct regs: loads clause
#pragma unroll
        for (int n8i = 0; n8i < 8; ++n8i) {
          int base = ((kk * 16 + half * 8 + n8i) * 32 + lane) * 2;
          bfr[n8i].u[0] = wl[base];
          bfr[n8i].u[1] = wl[base + 1];
        }
#pragma unroll
        for (int n8i = 0; n8i < 8; ++n8i)           // 8 back-to-back WMMAs
          acc[n8i] = __builtin_amdgcn_wmma_f32_16x16x32_bf16(
              false, afr[kk].v, false, bfr[n8i].v, (short)0, acc[n8i],
              false, false);
      }
      // batched convert + transposed store to zs
      int jb = (h << 3) + (half << 4);              // j' = row + 16*branch
#pragma unroll
      for (int n8i = 0; n8i < 8; ++n8i) {
        int d = (n8i << 4) + m;                     // feature row in zs
        uint4 pk;
        pk.x = pk2(acc[n8i][0], acc[n8i][1]);
        pk.y = pk2(acc[n8i][2], acc[n8i][3]);
        pk.z = pk2(acc[n8i][4], acc[n8i][5]);
        pk.w = pk2(acc[n8i][6], acc[n8i][7]);
        *(uint4*)&zs[d * 32 + jb] = pk;
      }
    }
    // --- aggregation (transposed): out^T = Zs^T-tile @ Mcat^T
    BF16x16 za[8];                                  // issue all loads first
#pragma unroll
    for (int dt = 0; dt < 8; ++dt) {
      const uint16_t* zrow = zs + (size_t)(dt * 16 + m) * 32;
      za[dt].u[0] = *(const uint4*)&zrow[h * 16];
      za[dt].u[1] = *(const uint4*)&zrow[h * 16 + 8];
    }
#pragma unroll
    for (int dt = 0; dt < 8; ++dt) {
      v8f cz = {0.f, 0.f, 0.f, 0.f, 0.f, 0.f, 0.f, 0.f};
      v8f o = __builtin_amdgcn_wmma_f32_16x16x32_bf16(
          false, za[dt].v, false, mct.v, (short)0, cz, false, false);
      uint4 pk;                                     // 8 consecutive features
      pk.x = pk2(o[0], o[1]);
      pk.y = pk2(o[2], o[3]);
      pk.z = pk2(o[4], o[5]);
      pk.w = pk2(o[6], o[7]);
      *(uint4*)&Y[(size_t)(b * 16 + m) * HIDC + dt * 16 + 8 * h] = pk;
    }
  }
}

// ---------------------------------------------------------------------------
// BN statistics pass: per-feature sum / sumsq of Y, LDS block reduction,
// then global atomics into stats[256].
// ---------------------------------------------------------------------------
__global__ void stats_kernel(const uint16_t* __restrict__ Y,
                             float* __restrict__ stats, long n8) {
  __shared__ float ls[256];
  int t = threadIdx.x;                               // 256 threads
  ls[t] = 0.0f;
  __syncthreads();
  float s[8], q[8];
#pragma unroll
  for (int e = 0; e < 8; ++e) { s[e] = 0.0f; q[e] = 0.0f; }
  const int d0 = (t & 15) * 8;                       // invariant per thread
  const long stride = (long)gridDim.x * blockDim.x;
  for (long i = (long)blockIdx.x * blockDim.x + t; i < n8; i += stride) {
    uint4 y = ((const uint4*)Y)[i];
    const uint16_t* ye = (const uint16_t*)&y;
#pragma unroll
    for (int e = 0; e < 8; ++e) {
      float f = bf2f(ye[e]);
      s[e] += f; q[e] += f * f;
    }
  }
#pragma unroll
  for (int e = 0; e < 8; ++e) {
    atomicAdd(&ls[d0 + e], s[e]);
    atomicAdd(&ls[128 + d0 + e], q[e]);
  }
  __syncthreads();
  atomicAdd(&stats[t], ls[t]);
}

// ---------------------------------------------------------------------------
// Finalize BN: scale/shift from accumulated sum/sumsq; re-zero stats.
// ---------------------------------------------------------------------------
__global__ void finalize_kernel(float* __restrict__ stats,
                                const float* __restrict__ gamma,
                                const float* __restrict__ beta,
                                float* __restrict__ ssb, float invN) {
  int d = threadIdx.x;                                         // 128
  float mean = stats[d] * invN;
  float var  = stats[128 + d] * invN - mean * mean;
  float sc   = gamma[d] * rsqrtf(var + EPSV);
  ssb[d] = sc;
  ssb[128 + d] = beta[d] - mean * sc;
  stats[d] = 0.0f;
  stats[128 + d] = 0.0f;
}

// ---------------------------------------------------------------------------
// BN + ReLU (+ optional residual add), 8 bf16 elements per thread.
// ---------------------------------------------------------------------------
template <bool RES>
__global__ void bn_kernel(const uint16_t* __restrict__ Yin,
                          const float* __restrict__ ssb,
                          uint16_t* __restrict__ act,
                          const uint16_t* __restrict__ resIn, long n8) {
  long i = (long)blockIdx.x * blockDim.x + threadIdx.x;
  if (i >= n8) return;
  int d0 = (int)((i * 8) & (HIDC - 1));
  uint4 y = ((const uint4*)Yin)[i];
  uint4 r;
  if (RES) r = ((const uint4*)resIn)[i];
  const uint16_t* ye = (const uint16_t*)&y;
  const uint16_t* re = (const uint16_t*)&r;
  float v[8];
#pragma unroll
  for (int e = 0; e < 8; ++e) {
    int d = d0 + e;
    v[e] = fmaxf(bf2f(ye[e]) * ssb[d] + ssb[128 + d], 0.0f);
    if (RES) v[e] += bf2f(re[e]);
  }
  uint4 o;
  o.x = pk2(v[0], v[1]);
  o.y = pk2(v[2], v[3]);
  o.z = pk2(v[4], v[5]);
  o.w = pk2(v[6], v[7]);
  ((uint4*)act)[i] = o;
}

// ---------------------------------------------------------------------------
// Output layer 128 -> 3 (~0.3% of FLOPs): one block per batch element.
// ---------------------------------------------------------------------------
__global__ __launch_bounds__(128) void out_kernel(
    const uint16_t* __restrict__ act, const float* __restrict__ W_out,
    const float* __restrict__ b_out, const float* __restrict__ mcat9,
    float* __restrict__ out, int B) {
  __shared__ float xt[JNT][HIDC];
  __shared__ float w[2 * HIDC * COUTC];
  __shared__ float z[32][COUTC];
  int b = blockIdx.x;
  if (b >= B) return;
  int t = threadIdx.x;
  for (int i = t; i < JNT * HIDC; i += 128)
    xt[i >> 7][i & 127] = bf2f(act[(size_t)b * JNT * HIDC + i]);
  for (int i = t; i < 2 * HIDC * COUTC; i += 128) w[i] = W_out[i];
  __syncthreads();
  if (t < 32 * COUTC) {
    int jp = t / COUTC, c = t % COUTC;
    int br = jp >> 4, j = jp & 15;
    float s = 0.0f;
    for (int k = 0; k < HIDC; ++k) s += xt[j][k] * w[br * (HIDC * COUTC) + k * COUTC + c];
    z[jp][c] = s;
  }
  __syncthreads();
  if (t < JNT * COUTC) {
    int i = t / COUTC, c = t % COUTC;
    float s = b_out[c];
    const float* mr = mcat9 + i * 32;
    for (int jp = 0; jp < 32; ++jp) s += mr[jp] * z[jp][c];
    out[((size_t)b * JNT + i) * COUTC + c] = s;
  }
}

// ---------------------------------------------------------------------------
extern "C" void kernel_launch(void* const* d_in, const int* in_sizes, int n_in,
                              void* d_out, int out_size, void* d_ws, size_t ws_size,
                              hipStream_t stream) {
  const float* x      = (const float*)d_in[0];
  const float* adj    = (const float*)d_in[1];
  const float* W_in   = (const float*)d_in[2];
  const float* e_in   = (const float*)d_in[4];
  const float* g_in   = (const float*)d_in[5];
  const float* bt_in  = (const float*)d_in[6];
  const float* W_res  = (const float*)d_in[7];
  const float* e_res  = (const float*)d_in[9];
  const float* g_res  = (const float*)d_in[10];
  const float* bt_res = (const float*)d_in[11];
  const float* W_out  = (const float*)d_in[12];
  const float* b_out  = (const float*)d_in[13];
  const float* e_out  = (const float*)d_in[14];
  // b_in (d_in[3]) and b_res (d_in[8]) cancel under training-mode BN: unused.
  float* out = (float*)d_out;
  const int B = in_sizes[0] / (JNT * CINC);

  char* ws = (char*)d_ws;
  float*    mcatRaw  = (float*)(ws + 0);           // 10*16*32*4    = 20480
  uint32_t* mcatT    = (uint32_t*)(ws + 20480);    // 10*32*8*4     = 10240
  uint32_t* pkIn     = (uint32_t*)(ws + 30720);    // 16*32*8*4     = 16384
  uint32_t* pkRes    = (uint32_t*)(ws + 47104);    // 8*4*16*32*8*4 = 524288
  float*    stats    = (float*)(ws + 571392);      // 256*4
  float*    ssb      = (float*)(ws + 572416);      // 256*4
  uint16_t* Xp       = (uint16_t*)(ws + 573440);   // B*16*32*2
  size_t actBytes = (size_t)B * JNT * HIDC * 2;
  uint16_t* act0 = (uint16_t*)(ws + 573440 + (size_t)B * JNT * 32 * 2);
  uint16_t* act1 = (uint16_t*)((char*)act0 + actBytes);
  uint16_t* act2 = (uint16_t*)((char*)act1 + actBytes);

  const int smemK1 = 1 * 16384 + 4 * 8192;   // 48 KB
  const int smemK4 = 4 * 16384 + 4 * 8192;   // 96 KB (CDNA5 WGP has 320 KB LDS)
  hipFuncSetAttribute(reinterpret_cast<const void*>(gemm_kernel<1>),
                      hipFuncAttributeMaxDynamicSharedMemorySize, smemK1);
  hipFuncSetAttribute(reinterpret_cast<const void*>(gemm_kernel<4>),
                      hipFuncAttributeMaxDynamicSharedMemorySize, smemK4);

  const float invN = 1.0f / ((float)B * JNT);
  const long n8 = (long)B * JNT * HIDC / 8;
  const int bnGrid = (int)((n8 + 255) / 256);
  const int gemmGrid = 512;
  const int statsGrid = 1024;

  // ---- prep / pack / pad -------------------------------------------------
  prep_kernel<<<1, 512, 0, stream>>>(adj, e_in, e_res, e_out, mcatRaw, mcatT, stats);
  pack_kernel<<<(8 * 4 * 16 * 32 + 16 * 32 + 255) / 256, 256, 0, stream>>>(
      W_in, W_res, pkIn, pkRes);
  pad_kernel<<<(B * 16 * 16 + 255) / 256, 256, 0, stream>>>(x, (uint32_t*)Xp,
                                                            B * 16 * 16);

  const uint4* mcT = (const uint4*)mcatT;

  // ---- input layer: gconv(5->128) + BN + ReLU ----------------------------
  gemm_kernel<1><<<gemmGrid, 128, smemK1, stream>>>(
      (const uint4*)Xp, (const uint4*)pkIn, mcT + 0 * 64, act0, B);
  stats_kernel<<<statsGrid, 256, 0, stream>>>(act0, stats, n8);
  finalize_kernel<<<1, 128, 0, stream>>>(stats, g_in, bt_in, ssb, invN);
  bn_kernel<false><<<bnGrid, 256, 0, stream>>>(act0, ssb, act0, nullptr, n8);

  // ---- 4 residual blocks (2 gconvs each) ---------------------------------
  for (int blk = 0; blk < 4; ++blk) {
    int a = 2 * blk, c = 2 * blk + 1;
    gemm_kernel<4><<<gemmGrid, 128, smemK4, stream>>>(
        (const uint4*)act0, (const uint4*)pkRes + (size_t)a * 4096,
        mcT + (size_t)(1 + a) * 64, act1, B);
    stats_kernel<<<statsGrid, 256, 0, stream>>>(act1, stats, n8);
    finalize_kernel<<<1, 128, 0, stream>>>(stats, g_res + a * HIDC,
                                           bt_res + a * HIDC, ssb, invN);
    bn_kernel<false><<<bnGrid, 256, 0, stream>>>(act1, ssb, act1, nullptr, n8);

    gemm_kernel<4><<<gemmGrid, 128, smemK4, stream>>>(
        (const uint4*)act1, (const uint4*)pkRes + (size_t)c * 4096,
        mcT + (size_t)(1 + c) * 64, act2, B);
    stats_kernel<<<statsGrid, 256, 0, stream>>>(act2, stats, n8);
    finalize_kernel<<<1, 128, 0, stream>>>(stats, g_res + c * HIDC,
                                           bt_res + c * HIDC, ssb, invN);
    bn_kernel<true><<<bnGrid, 256, 0, stream>>>(act2, ssb, act0, act0, n8);
  }

  // ---- output layer 128 -> 3 --------------------------------------------
  out_kernel<<<B, 128, 0, stream>>>(act0, W_out, b_out, mcatRaw + 9 * JNT * 32,
                                    out, B);
}
